// SSIR_STGCN_87935160418920
// MI455X (gfx1250) — matrix operand order repaired and checked by reading, not compiled, not run on hardware
//
#include <hip/hip_runtime.h>
#include <hip/hip_bf16.h>
#include <math.h>

typedef __attribute__((ext_vector_type(16))) _Float16 v16h;
typedef __attribute__((ext_vector_type(8)))  _Float16 v8h;
typedef __attribute__((ext_vector_type(8)))  float    v8f;

#define HIDN  64
#define PRE   14
#define NBAT  4
#define TT    12
#define NNODE 128
#define FFEAT 3
#define KPAD  96      // padded K (F+H=67 or 2F+H=70 -> 96 = 3 chunks of 32)
#define GATES 256     // 4*H

__device__ __forceinline__ float sigmoidf_(float x) {
  return 1.0f / (1.0f + __expf(-x));
}

// ---------------------------------------------------------------------------
// Build f16 combined+padded weight matrices Wt[which][n][k]:
//   which 0: beta   (k<3 : bWih, 3..66 : bWhh, else 0)
//   which 1: gamma  (same with g*)
//   which 2: edge   (k<6 : cWih, 6..69 : cWhh, else 0)
// Stored row-major [256][96] per LSTM so a WMMA B-tile lane reads 32B runs.
// ---------------------------------------------------------------------------
__global__ void prep_weights(const float* bWih, const float* bWhh,
                             const float* gWih, const float* gWhh,
                             const float* cWih, const float* cWhh,
                             _Float16* Wt) {
  int r = blockIdx.x;            // 0..767
  int which = r >> 8;
  int n = r & 255;
  int k = threadIdx.x;           // 0..95
  float v = 0.0f;
  if (which == 0) {
    if (k < 3)       v = bWih[n * 3 + k];
    else if (k < 67) v = bWhh[n * 64 + (k - 3)];
  } else if (which == 1) {
    if (k < 3)       v = gWih[n * 3 + k];
    else if (k < 67) v = gWhh[n * 64 + (k - 3)];
  } else {
    if (k < 6)       v = cWih[n * 6 + k];
    else if (k < 70) v = cWhh[n * 64 + (k - 6)];
  }
  Wt[(which * 256 + n) * KPAD + k] = (_Float16)v;
}

// ---------------------------------------------------------------------------
// LSTM over a 16-sequence M-tile, full 12-step recurrence in one workgroup.
// 8 waves; each wave owns 2 N-tiles (32 of 256 gate outputs) and keeps its
// 6 v16h weight B-tiles in VGPRs for the whole kernel.
// EDGE=0: node LSTM (beta/gamma), 3 input features, seq = b*N + n.
// EDGE=1: edge LSTM, 6 input features, seq = (b, i, j0+m).
// ---------------------------------------------------------------------------
template <int EDGE>
__global__ __launch_bounds__(256)
void lstm_tile(const float* __restrict__ xs,   // x_s (B,T,N,F)
               const _Float16* __restrict__ Wt,
               const float* __restrict__ bias, // 256
               const float* __restrict__ fW,   // 14*64
               const float* __restrict__ fb,   // 14
               float* __restrict__ epi,        // rows of 130
               int slot)
{
  __shared__ __align__(16) _Float16 Apad[16 * KPAD]; // [xt | h | pad] f16
  __shared__ float gLDS[16 * GATES];
  __shared__ float cLDS[16 * HIDN];
  __shared__ float hLDS[16 * HIDN];

  const int tid  = threadIdx.x;
  const int wave = tid >> 5;
  const int lane = tid & 31;
  const int tile = blockIdx.x;

  int b, i, j0, n0;
  if (EDGE) {
    b  = tile >> 10;               // 1024 tiles per batch
    int rem = tile & 1023;
    i  = rem >> 3;                 // 8 tiles per i
    j0 = (rem & 7) << 4;
    n0 = 0;
  } else {
    b  = tile >> 3;                // 8 tiles per batch (128/16)
    n0 = (tile & 7) << 4;
    i = 0; j0 = 0;
  }
  const int XOFF = EDGE ? 6 : 3;   // column where h starts

  // init: A buffer zero (pad cols + h0=0), c0 = 0
  for (int idx = tid; idx < 16 * KPAD; idx += 256) Apad[idx] = (_Float16)0.0f;
  for (int idx = tid; idx < 16 * HIDN; idx += 256) cLDS[idx] = 0.0f;

  // ---- resident WMMA B-tiles: lane holds col=lane&15, K = kb..kb+15
  const int col = lane & 15;
  const int bkb = (lane < 16) ? 0 : 16;
  v16h wB[2][3];
  float bval[2];
#pragma unroll
  for (int s = 0; s < 2; ++s) {
    const int nt = wave * 2 + s;
#pragma unroll
    for (int c = 0; c < 3; ++c)
      wB[s][c] = *(const v16h*)(Wt + (nt * 16 + col) * KPAD + c * 32 + bkb);
    bval[s] = bias[nt * 16 + col];
  }

  // A-layout gather params: lane holds row=lane&15,
  // halves K = akb..akb+15 within each K=32 chunk
  const int arow = lane & 15;
  const int akb  = (lane < 16) ? 0 : 16;

  for (int t = 0; t < TT; ++t) {
    // ---- stage xt into Apad
    if (tid < 16) {
      const int m = tid;
      if (EDGE) {
        const float* xi = xs + ((size_t)(b * TT + t) * NNODE + i) * FFEAT;
        const float* xj = xs + ((size_t)(b * TT + t) * NNODE + (j0 + m)) * FFEAT;
        Apad[m * KPAD + 0] = (_Float16)xi[0];
        Apad[m * KPAD + 1] = (_Float16)xi[1];
        Apad[m * KPAD + 2] = (_Float16)xi[2];
        Apad[m * KPAD + 3] = (_Float16)xj[0];
        Apad[m * KPAD + 4] = (_Float16)xj[1];
        Apad[m * KPAD + 5] = (_Float16)xj[2];
        if (t + 1 < TT)    // warm L2/GL1 for next step's x rows
          __builtin_prefetch(xs + ((size_t)(b * TT + t + 1) * NNODE + (j0 + m)) * FFEAT, 0, 0);
      } else {
        const float* xp = xs + ((size_t)(b * TT + t) * NNODE + (n0 + m)) * FFEAT;
        Apad[m * KPAD + 0] = (_Float16)xp[0];
        Apad[m * KPAD + 1] = (_Float16)xp[1];
        Apad[m * KPAD + 2] = (_Float16)xp[2];
      }
    }
    __syncthreads();

    // ---- gates: g = [xt|h] @ Wcomb.T + b via 6 WMMAs per wave
#pragma unroll
    for (int s = 0; s < 2; ++s) {
      v8f acc;
#pragma unroll
      for (int e = 0; e < 8; ++e) acc[e] = bval[s];   // bias as C (same n per lane)
#pragma unroll
      for (int c = 0; c < 3; ++c) {
        const v8h lo = *(const v8h*)(Apad + arow * KPAD + c * 32 + akb);
        const v8h hi = *(const v8h*)(Apad + arow * KPAD + c * 32 + akb + 8);
        v16h a;
#pragma unroll
        for (int j = 0; j < 8; ++j) { a[j] = lo[j]; a[8 + j] = hi[j]; }
        acc = __builtin_amdgcn_wmma_f32_16x16x32_f16(false, a, false, wB[s][c],
                                                     (short)0, acc, false, false);
      }
      const int nt   = wave * 2 + s;
      const int moff = (lane < 16) ? 0 : 8;           // D: VGPR v -> row v(+8)
#pragma unroll
      for (int v = 0; v < 8; ++v)
        gLDS[(v + moff) * GATES + nt * 16 + col] = acc[v];
    }
    __syncthreads();

    // ---- LSTM cell nonlinearity: 16x64 states, 4 per thread
#pragma unroll
    for (int q = 0; q < 4; ++q) {
      const int idx = tid + q * 256;
      const int m = idx >> 6;
      const int h = idx & 63;
      const float gi = gLDS[m * GATES + h];
      const float gf = gLDS[m * GATES + 64 + h];
      const float gg = gLDS[m * GATES + 128 + h];
      const float go = gLDS[m * GATES + 192 + h];
      const float cc = sigmoidf_(gf) * cLDS[idx] + sigmoidf_(gi) * tanhf(gg);
      const float hh = sigmoidf_(go) * tanhf(cc);
      cLDS[idx] = cc;
      hLDS[idx] = hh;
      Apad[m * KPAD + XOFF + h] = (_Float16)hh;
    }
    __syncthreads();
  }

  // ---- FC head + sigmoid, scatter into epi rows
  if (tid < 16 * PRE) {
    const int m = tid / PRE;
    const int p = tid % PRE;
    float acc = fb[p];
    for (int k = 0; k < HIDN; ++k) acc += hLDS[m * HIDN + k] * fW[p * HIDN + k];
    const float val = sigmoidf_(acc);
    if (EDGE) {
      epi[((size_t)(b * PRE + p) * NNODE + i) * 130 + 2 + (j0 + m)] = val;
    } else {
      const int n = n0 + m;
      epi[((size_t)(b * PRE + p) * NNODE + n) * 130 + slot] = val;
    }
  }
}

// ---------------------------------------------------------------------------
// In-place softmax over j for each (b,p,i) row of epi (128 contiguous floats).
// ---------------------------------------------------------------------------
__global__ __launch_bounds__(128)
void softmax_rows(float* __restrict__ epi) {
  __shared__ float red[128];
  float* ptr = epi + (size_t)blockIdx.x * 130 + 2;
  const int tid = threadIdx.x;
  const float v = ptr[tid];
  red[tid] = v;
  __syncthreads();
  for (int s = 64; s > 0; s >>= 1) {
    if (tid < s) red[tid] = fmaxf(red[tid], red[tid + s]);
    __syncthreads();
  }
  const float mx = red[0];
  __syncthreads();
  const float e = __expf(v - mx);
  red[tid] = e;
  __syncthreads();
  for (int s = 64; s > 0; s >>= 1) {
    if (tid < s) red[tid] += red[tid + s];
    __syncthreads();
  }
  ptr[tid] = e / red[0];
}

// ---------------------------------------------------------------------------
// SIR scan: 14 sequential steps, infection = c @ I per step; c/beta/gamma
// read back from the epi output region. One block per batch, thread = node i.
// ---------------------------------------------------------------------------
__global__ __launch_bounds__(128)
void sir_scan(const float* __restrict__ x, const float* __restrict__ epi,
              float* __restrict__ out1, float* __restrict__ out3,
              float* __restrict__ out4) {
  __shared__ float Ssh[NNODE], Ish[NNODE], Rsh[NNODE];
  const int b = blockIdx.x;
  const int i = threadIdx.x;
  const float* x0 = x + ((size_t)(b * TT + (TT - 1)) * NNODE + i) * FFEAT;
  Ssh[i] = x0[0]; Ish[i] = x0[1]; Rsh[i] = x0[2];
  const float eps = 1e-8f;
  for (int p = 0; p < PRE; ++p) {
    __syncthreads();
    const float* erow = epi + ((size_t)(b * PRE + p) * NNODE + i) * 130;
    const float beta  = erow[0];
    const float gamma = erow[1];
    const float Si = Ssh[i], Ii = Ish[i], Ri = Rsh[i];
    const float Ntot = fmaxf(Si + Ii + Ri, eps);
    float inf = 0.0f;
    for (int j = 0; j < NNODE; ++j) inf += erow[2 + j] * Ish[j];
    const float dS = -beta * Si / Ntot * inf;
    const float dI = -dS - gamma * Ii;
    const float dR = gamma * Ii;
    const float St = fmaxf(Si + dS, 0.0f);
    const float It = fmaxf(Ii + dI, 0.0f);
    const float Rt = fmaxf(Ri + dR, 0.0f);
    const float scale = Ntot / fmaxf(St + It + Rt, eps);
    const float Inew = fmaxf(-dS, 0.0f);
    const float Ss = St * scale, Is = It * scale, Rs = Rt * scale;
    const size_t o = (size_t)(b * PRE + p) * NNODE + i;
    out1[o] = Is;
    out3[o * 3 + 0] = Ss;
    out3[o * 3 + 1] = Is;
    out3[o * 3 + 2] = Rs;
    out4[o] = Inew;
    __syncthreads();
    Ssh[i] = Ss; Ish[i] = Is; Rsh[i] = Rs;
  }
}

extern "C" void kernel_launch(void* const* d_in, const int* in_sizes, int n_in,
                              void* d_out, int out_size, void* d_ws, size_t ws_size,
                              hipStream_t stream) {
  const float* x_s  = (const float*)d_in[0];
  const float* x    = (const float*)d_in[1];
  const float* bWih = (const float*)d_in[2];
  const float* bWhh = (const float*)d_in[3];
  const float* bb   = (const float*)d_in[4];
  const float* bfW  = (const float*)d_in[5];
  const float* bfb  = (const float*)d_in[6];
  const float* gWih = (const float*)d_in[7];
  const float* gWhh = (const float*)d_in[8];
  const float* gb   = (const float*)d_in[9];
  const float* gfW  = (const float*)d_in[10];
  const float* gfb  = (const float*)d_in[11];
  const float* cWih = (const float*)d_in[12];
  const float* cWhh = (const float*)d_in[13];
  const float* cb   = (const float*)d_in[14];
  const float* cfW  = (const float*)d_in[15];
  const float* cfb  = (const float*)d_in[16];

  // output layout (flat, return order): out1 | epi | out3 | out4
  float* out  = (float*)d_out;
  float* out1 = out;                                         // B*P*N
  float* epi  = out1 + (size_t)NBAT * PRE * NNODE;           // B*P*N*130
  float* out3 = epi  + (size_t)NBAT * PRE * NNODE * 130;     // B*P*N*3
  float* out4 = out3 + (size_t)NBAT * PRE * NNODE * 3;       // B*P*N

  _Float16* Wt = (_Float16*)d_ws;  // 3 * 256 * 96 halves = 147 KB

  prep_weights<<<768, 96, 0, stream>>>(bWih, bWhh, gWih, gWhh, cWih, cWhh, Wt);

  lstm_tile<0><<<32, 256, 0, stream>>>(x_s, Wt, bb, bfW, bfb, epi, 0);
  lstm_tile<0><<<32, 256, 0, stream>>>(x_s, Wt + 256 * KPAD, gb, gfW, gfb, epi, 1);
  lstm_tile<1><<<4096, 256, 0, stream>>>(x_s, Wt + 2 * 256 * KPAD, cb, cfW, cfb, epi, 0);

  softmax_rows<<<NBAT * PRE * NNODE, 128, 0, stream>>>(epi);
  sir_scan<<<NBAT, 128, 0, stream>>>(x, epi, out1, out3, out4);
}
